// GroupedQueryAttentionHead_64759516889367
// MI455X (gfx1250) — matrix operand order repaired and checked by reading, not compiled, not run on hardware
//
#include <hip/hip_runtime.h>

// ---------------------------------------------------------------------------
// GQA head for MI455X (gfx1250): bf16 WMMA (v_wmma_f32_16x16x32_bf16) with
// f32 accumulation, flash-attention online softmax (base-2), async global->LDS
// staging for V (ASYNCcnt) and Tensor Data Mover staging for K (TENSORcnt).
// ---------------------------------------------------------------------------

typedef __attribute__((ext_vector_type(16))) __bf16    v16bf;
typedef __attribute__((ext_vector_type(8)))  __bf16    v8bf;
typedef __attribute__((ext_vector_type(8)))  float     v8f;
typedef __attribute__((ext_vector_type(4)))  unsigned  v4u;
typedef __attribute__((ext_vector_type(8)))  int       v8i;
typedef __attribute__((ext_vector_type(4)))  int       v4i;

#define EMB  2048
#define DH   128
#define SEQL 4096
#define BSZ  4

static __device__ __forceinline__ v8f wmma_bf16(v16bf a, v16bf b, v8f c) {
    // (neg_a, A, neg_b, B, c_mod, C, reuse_a, reuse_b)
    return __builtin_amdgcn_wmma_f32_16x16x32_bf16(false, a, false, b,
                                                   (short)0, c, false, false);
}

static __device__ __forceinline__ void async_copy16(unsigned lds_addr,
                                                    const void* gptr) {
    asm volatile("global_load_async_to_lds_b128 %0, %1, off"
                 :: "v"(lds_addr), "v"((unsigned long long)(size_t)gptr)
                 : "memory");
}

// TDM: 2D tile load global -> LDS with hardware row padding (TENSORcnt).
//   w0_flags: data_size / pad_enable / pad_interval / pad_amount bits of D# G1.
static __device__ __forceinline__ void tdm_load_2d(
    unsigned lds_addr, const void* gaddr, unsigned w0_flags,
    unsigned tensor_d0, unsigned tensor_d1,
    unsigned tile_d0, unsigned tile_d1, unsigned stride0)
{
    const unsigned long long ga = (unsigned long long)(size_t)gaddr;
    v4u g0;
    g0[0] = 1u;                                              // count=1 (valid)
    g0[1] = lds_addr;                                        // LDS byte address
    g0[2] = (unsigned)(ga & 0xFFFFFFFFu);                    // global_addr lo
    g0[3] = (unsigned)((ga >> 32) & 0x01FFFFFFu) | 0x80000000u; // hi | type=2
    v8i g1;
    g1[0] = (int)w0_flags;                                   // wg_mask=0 | cfg
    g1[1] = (int)(tensor_d0 << 16);                          // tensor_dim0 lo
    g1[2] = (int)((tensor_d0 >> 16) | (tensor_d1 << 16));    // td0 hi | td1 lo
    g1[3] = (int)((tensor_d1 >> 16) | (tile_d0 << 16));      // td1 hi | tile_d0
    g1[4] = (int)(tile_d1 & 0xFFFFu);                        // tile_d1, tile_d2=0
    g1[5] = (int)stride0;                                    // dim0 stride lo
    g1[6] = 0;
    g1[7] = 0;
    const v4i z4 = {0, 0, 0, 0};
    const v8i z8 = {0, 0, 0, 0, 0, 0, 0, 0};
    // 6-arg toolchain form: (g0, g1, g2, g3, g4, cpol)
    __builtin_amdgcn_tensor_load_to_lds(g0, g1, z4, z4, z8, 0);
}

// data_size=2B | pad_enable | pad every 64 dwords (256B rows) by 4 dwords
#define TDM_FLAGS_K ((1u << 16) | (1u << 20) | (5u << 22) | (3u << 25))

// ---------------------------------------------------------------------------
// Kernel 0: W [2048][128] f32  ->  Wt [128][2048] bf16 (transpose + convert)
// ---------------------------------------------------------------------------
__global__ __launch_bounds__(256) void wcvt_kernel(const float* __restrict__ W,
                                                   __bf16* __restrict__ Wt) {
    int idx = blockIdx.x * 256 + threadIdx.x;   // 0 .. 2048*128-1
    int k = idx >> 7;
    int n = idx & 127;
    Wt[(size_t)n * EMB + k] = (__bf16)W[idx];
}

// ---------------------------------------------------------------------------
// Kernel 1: QKV projection + RoPE.
//   Q -> Qrm [B*S][128] bf16, K -> Krm [B*S][128] bf16, V -> Vt [B][128][S] bf16
// ---------------------------------------------------------------------------
__global__ __launch_bounds__(256) void qkv_rope_kernel(
    const float* __restrict__ x,
    const float* __restrict__ sinT, const float* __restrict__ cosT,
    const __bf16* __restrict__ WtQ, const __bf16* __restrict__ WtK,
    const __bf16* __restrict__ WtV,
    __bf16* __restrict__ Qrm, __bf16* __restrict__ Krm, __bf16* __restrict__ Vt)
{
    __shared__ __align__(16) __bf16 xs[16 * 136];   // 16 x 128, stride 136 (pad)

    const int tid  = threadIdx.x;
    const int lane = tid & 31;
    const int wave = tid >> 5;
    const int half = lane >> 4;
    const int ln   = lane & 15;

    const int m0 = blockIdx.x * 16;     // flattened row base over [B*S]
    const int b  = m0 >> 12;
    const int s0 = m0 & 4095;
    const int colbase = wave * 16;

    v8f accQ = {}; v8f accK = {}; v8f accV = {};

    const int frow = tid >> 4;          // fill: row 0..15
    const int fcol = (tid & 15) << 2;   // fill: col 0..60 step 4

    for (int kc = 0; kc < EMB; kc += 128) {
        #pragma unroll
        for (int rep = 0; rep < 2; ++rep) {
            int col = fcol + rep * 64;
            float4 v = *(const float4*)(x + (size_t)(m0 + frow) * EMB + kc + col);
            __bf16* dst = &xs[frow * 136 + col];
            dst[0] = (__bf16)v.x; dst[1] = (__bf16)v.y;
            dst[2] = (__bf16)v.z; dst[3] = (__bf16)v.w;
        }
        __syncthreads();

        #pragma unroll
        for (int sub = 0; sub < 4; ++sub) {
            v16bf a;
            const __bf16* ap = &xs[ln * 136 + sub * 32 + half * 16];
            *((v8bf*)&a)     = *(const v8bf*)ap;          // K + 0..7
            *((v8bf*)&a + 1) = *(const v8bf*)(ap + 8);    // K + 8..15

            const int kk = kc + sub * 32 + half * 16;
            const size_t wofs = (size_t)(colbase + ln) * EMB + kk;

            v16bf bq, bk, bv;
            const __bf16* qp = WtQ + wofs;
            *((v8bf*)&bq) = *(const v8bf*)qp;  *((v8bf*)&bq + 1) = *(const v8bf*)(qp + 8);
            const __bf16* kp = WtK + wofs;
            *((v8bf*)&bk) = *(const v8bf*)kp;  *((v8bf*)&bk + 1) = *(const v8bf*)(kp + 8);
            const __bf16* vp = WtV + wofs;
            *((v8bf*)&bv) = *(const v8bf*)vp;  *((v8bf*)&bv + 1) = *(const v8bf*)(vp + 8);

            accQ = wmma_bf16(a, bq, accQ);
            accK = wmma_bf16(a, bk, accK);
            accV = wmma_bf16(a, bv, accV);
        }
        __syncthreads();
    }

    const int d = colbase + ln;
    const int j = d >> 1;
    #pragma unroll
    for (int r = 0; r < 8; ++r) {
        const int s = s0 + half * 8 + r;
        const float cv = cosT[s * 64 + j];
        const float sv = sinT[s * 64 + j];

        float q  = accQ[r];
        float k  = accK[r];
        float qp = __shfl_xor(q, 1, 32);
        float kp = __shfl_xor(k, 1, 32);
        float qr, kr;
        if ((ln & 1) == 0) { qr = q * cv - qp * sv;  kr = k * cv - kp * sv; }
        else               { qr = q * cv + qp * sv;  kr = k * cv + kp * sv; }

        const size_t rowoff = (size_t)(m0 + half * 8 + r) * DH + d;
        Qrm[rowoff] = (__bf16)qr;
        Krm[rowoff] = (__bf16)kr;
        Vt[(size_t)(b * DH + d) * SEQL + s] = (__bf16)accV[r];
    }
}

// ---------------------------------------------------------------------------
// Kernel 2: causal flash attention.
// 128-thread block (4 waves) owns 64 queries. Per 32-key step: K tile staged
// by TDM (wave 0, hardware-padded rows), V tile staged by per-lane async
// copies; all 4 waves consume the shared LDS tiles.
// ---------------------------------------------------------------------------
__global__ __launch_bounds__(128) void attn_kernel(
    const __bf16* __restrict__ Qrm, const __bf16* __restrict__ Krm,
    const __bf16* __restrict__ Vt,  float* __restrict__ out)
{
    __shared__ __align__(16) __bf16 Ks[32 * 136];      // 32 keys x 128 d (pad 8)
    __shared__ __align__(16) __bf16 Vs[128 * 40];      // 128 d x 32 seq (pad 8)
    __shared__ __align__(16) __bf16 Ps[4][16 * 40];    // per-wave P tile

    const int tid  = threadIdx.x;
    const int wave = tid >> 5;
    const int lane = tid & 31;
    const int half = lane >> 4;
    const int ln   = lane & 15;

    const int b     = blockIdx.x >> 6;                 // 64 blocks per batch
    const int qb0   = (blockIdx.x & 63) * 64;
    const int qbase = qb0 + wave * 16;

    const __bf16* Kg = Krm + (size_t)b * SEQL * DH;    // [4096][128]
    const __bf16* Vg = Vt  + (size_t)b * DH * SEQL;    // [128][4096]

    v16bf qf[4];
    const __bf16* qrow = Qrm + (size_t)(b * SEQL + qbase + ln) * DH;
    #pragma unroll
    for (int c = 0; c < 4; ++c) {
        const __bf16* p = qrow + c * 32 + half * 16;
        *((v8bf*)&qf[c])     = *(const v8bf*)p;
        *((v8bf*)&qf[c] + 1) = *(const v8bf*)(p + 8);
    }

    float m[8], l[8];
    v8f O[8];
    #pragma unroll
    for (int r = 0; r < 8; ++r) { m[r] = -__builtin_inff(); l[r] = 0.f; }
    #pragma unroll
    for (int t = 0; t < 8; ++t) O[t] = (v8f){};

    // base-2 softmax: fold log2(e) into the 1/sqrt(d_h) scale (v_exp_f32 is 2^x)
    const float scale = 0.08838834764831845f * 1.4426950408889634f;
    const int kend_blk = qb0 + 64;
    const int kend_w   = qbase + 16;

    for (int kb = 0; kb < kend_blk; kb += 32) {
        __syncthreads();   // previous step's LDS reads complete -> safe to refill

        // ---- K tile via Tensor Data Mover (wave 0; EXEC ignored by TDM) ----
        if (wave == 0) {
            tdm_load_2d((unsigned)(size_t)&Ks[0],
                        Kg + (size_t)kb * DH, TDM_FLAGS_K,
                        /*tensor_d0=*/DH, /*tensor_d1=*/SEQL,
                        /*tile_d0=*/DH, /*tile_d1=*/32, /*stride0=*/DH);
        }
        // ---- V tile (128 x 32) via per-lane async copies ----
        #pragma unroll
        for (int i = 0; i < 4; ++i) {
            const int L = tid + 128 * i;               // 0..511
            const int row = L >> 2, seg = L & 3;       // 128 rows x 4 segs
            async_copy16((unsigned)(size_t)&Vs[row * 40 + seg * 8],
                         Vg + (size_t)row * SEQL + kb + seg * 8);
        }
        asm volatile("s_wait_asynccnt 0" ::: "memory");
        if (wave == 0) __builtin_amdgcn_s_wait_tensorcnt(0);
        __syncthreads();   // staged tiles visible to all waves

        if (kb < kend_w) {  // uniform per wave: EXEC stays all-ones inside
            v8f S0 = (v8f){}, S1 = (v8f){};
            #pragma unroll
            for (int c = 0; c < 4; ++c) {
                v16bf b0, b1;
                const __bf16* k0p = &Ks[ln * 136 + c * 32 + half * 16];
                *((v8bf*)&b0) = *(const v8bf*)k0p;
                *((v8bf*)&b0 + 1) = *(const v8bf*)(k0p + 8);
                const __bf16* k1p = k0p + 16 * 136;
                *((v8bf*)&b1) = *(const v8bf*)k1p;
                *((v8bf*)&b1 + 1) = *(const v8bf*)(k1p + 8);
                S0 = wmma_bf16(qf[c], b0, S0);
                S1 = wmma_bf16(qf[c], b1, S1);
            }

            __bf16* pw = &Ps[wave][0];
            #pragma unroll
            for (int r = 0; r < 8; ++r) {
                const int q = qbase + half * 8 + r;
                float s0v = (kb + ln      <= q) ? S0[r] * scale : -__builtin_inff();
                float s1v = (kb + 16 + ln <= q) ? S1[r] * scale : -__builtin_inff();

                float rm = fmaxf(s0v, s1v);
                #pragma unroll
                for (int off = 1; off < 16; off <<= 1)
                    rm = fmaxf(rm, __shfl_xor(rm, off, 32));
                const float mnew  = fmaxf(m[r], rm);
                const float alpha = exp2f(m[r] - mnew);

                const float p0 = exp2f(s0v - mnew);
                const float p1 = exp2f(s1v - mnew);
                float rs = p0 + p1;
                #pragma unroll
                for (int off = 1; off < 16; off <<= 1)
                    rs += __shfl_xor(rs, off, 32);

                l[r] = l[r] * alpha + rs;
                m[r] = mnew;
                #pragma unroll
                for (int t = 0; t < 8; ++t) O[t][r] *= alpha;

                pw[(half * 8 + r) * 40 + ln]      = (__bf16)p0;
                pw[(half * 8 + r) * 40 + 16 + ln] = (__bf16)p1;
            }
            // wave-local C->A re-layout: DS ops are in-order within a wave.
            asm volatile("" ::: "memory");

            v16bf pf;
            const __bf16* pp = &pw[ln * 40 + half * 16];
            *((v8bf*)&pf)     = *(const v8bf*)pp;
            *((v8bf*)&pf + 1) = *(const v8bf*)(pp + 8);

            #pragma unroll
            for (int t = 0; t < 8; ++t) {
                v16bf vf;
                const __bf16* vp = &Vs[(t * 16 + ln) * 40 + half * 16];
                *((v8bf*)&vf)     = *(const v8bf*)vp;
                *((v8bf*)&vf + 1) = *(const v8bf*)(vp + 8);
                O[t] = wmma_bf16(pf, vf, O[t]);
            }
        }
    }

    #pragma unroll
    for (int t = 0; t < 8; ++t)
        #pragma unroll
        for (int r = 0; r < 8; ++r) {
            const size_t o = (size_t)(b * SEQL + qbase + half * 8 + r) * DH + t * 16 + ln;
            out[o] = O[t][r] / l[r];
        }
}

// ---------------------------------------------------------------------------
// Launch
// ---------------------------------------------------------------------------
extern "C" void kernel_launch(void* const* d_in, const int* in_sizes, int n_in,
                              void* d_out, int out_size, void* d_ws, size_t ws_size,
                              hipStream_t stream) {
    const float* x    = (const float*)d_in[0];
    // d_in[1] = mask (unused; causal mask computed analytically)
    const float* sinT = (const float*)d_in[2];
    const float* cosT = (const float*)d_in[3];
    const float* WQ   = (const float*)d_in[4];
    const float* WK   = (const float*)d_in[5];
    const float* WV   = (const float*)d_in[6];
    float* out = (float*)d_out;

    __bf16* WtQ = (__bf16*)d_ws;                 // [128][2048]
    __bf16* WtK = WtQ + (size_t)DH * EMB;
    __bf16* WtV = WtK + (size_t)DH * EMB;
    __bf16* Qrm = WtV + (size_t)DH * EMB;        // [B*S][128]
    __bf16* Krm = Qrm + (size_t)BSZ * SEQL * DH;
    __bf16* Vt  = Krm + (size_t)BSZ * SEQL * DH; // [B][128][S]

    wcvt_kernel<<<(EMB * DH) / 256, 256, 0, stream>>>(WQ, WtQ);
    wcvt_kernel<<<(EMB * DH) / 256, 256, 0, stream>>>(WK, WtK);
    wcvt_kernel<<<(EMB * DH) / 256, 256, 0, stream>>>(WV, WtV);

    qkv_rope_kernel<<<(BSZ * SEQL) / 16, 256, 0, stream>>>(
        x, sinT, cosT, WtQ, WtK, WtV, Qrm, Krm, Vt);

    attn_kernel<<<BSZ * (SEQL / 64), 128, 0, stream>>>(Qrm, Krm, Vt, out);
}
